// KNN_4466765988030
// MI455X (gfx1250) — matrix operand order repaired and checked by reading, not compiled, not run on hardware
//
#include <hip/hip_runtime.h>

// Problem constants (match reference setup_inputs)
#define Q_TOTAL 4096
#define N_DB    65536
#define D_DIM   512
#define C_DIM   128
#define KNN_K   16

#define WAVES_PER_WG 4
#define Q_PER_WAVE   16
#define Q_PER_WG     (WAVES_PER_WG * Q_PER_WAVE)   // 64
#define TILE_M       16
#define NTILES       (N_DB / TILE_M)               // 4096
#define NSLICES      (D_DIM / 32)                  // 16 K-slices of 32
#define NSPLIT       8                              // N-dimension split for occupancy
#define TPS          (NTILES / NSPLIT)              // 512 tiles per segment
#define TILE_BYTES   32768                          // bf16 hi+lo staged tile
#define CHUNKS_PER_TILE 2048                        // 16B chunks per tile

typedef __attribute__((ext_vector_type(16))) __bf16 v16bf;
typedef __attribute__((ext_vector_type(8)))  float  v8f;

union V16 { v16bf v; uint4 u[2]; };

__device__ __forceinline__ unsigned short bf16_rne(float f) {
    unsigned u = __float_as_uint(f);
    return (unsigned short)((u + 0x7FFFu + ((u >> 16) & 1u)) >> 16);
}
__device__ __forceinline__ float bf16_tof(unsigned short h) {
    return __uint_as_float(((unsigned)h) << 16);
}
// Split 8 fp32 into packed bf16 hi and bf16 lo (residual) chunks.
__device__ __forceinline__ void split8(const float f[8], uint4& hi, uint4& lo) {
    unsigned short h[8], l[8];
    #pragma unroll
    for (int i = 0; i < 8; ++i) {
        h[i] = bf16_rne(f[i]);
        l[i] = bf16_rne(f[i] - bf16_tof(h[i]));
    }
    hi = make_uint4((unsigned)h[0] | ((unsigned)h[1] << 16),
                    (unsigned)h[2] | ((unsigned)h[3] << 16),
                    (unsigned)h[4] | ((unsigned)h[5] << 16),
                    (unsigned)h[6] | ((unsigned)h[7] << 16));
    lo = make_uint4((unsigned)l[0] | ((unsigned)l[1] << 16),
                    (unsigned)l[2] | ((unsigned)l[3] << 16),
                    (unsigned)l[4] | ((unsigned)l[5] << 16),
                    (unsigned)l[6] | ((unsigned)l[7] << 16));
}

// CDNA5 async memory->LDS copy (ASYNCcnt-tracked), per CDNA5 ISA.
__device__ __forceinline__ void async_copy_b128(unsigned lds_off, unsigned voff,
                                                const void* sbase) {
    asm volatile("global_load_async_to_lds_b128 %0, %1, %2 offset:0"
                 :: "v"(lds_off), "v"(voff), "s"(sbase) : "memory");
}
__device__ __forceinline__ void wait_async0() {
    asm volatile("s_wait_asynccnt 0x0" ::: "memory");
}

// ---------------------------------------------------------------------------
// Kernel 1: ||d||^2 per database row. Wave-per-row, coalesced float4 loads.
// ---------------------------------------------------------------------------
__global__ __launch_bounds__(256) void dbnorm_kernel(const float* __restrict__ db,
                                                     float* __restrict__ dsq) {
    const int wave = threadIdx.x >> 5, ln = threadIdx.x & 31;
    const int n = blockIdx.x * 8 + wave;
    const float4* row = (const float4*)(db + (size_t)n * D_DIM);
    float s = 0.f;
    #pragma unroll
    for (int j = 0; j < 4; ++j) {
        float4 v = row[ln + 32 * j];
        s += v.x * v.x + v.y * v.y + v.z * v.z + v.w * v.w;
    }
    #pragma unroll
    for (int off = 16; off > 0; off >>= 1) s += __shfl_xor(s, off, 32);
    if (ln == 0) dsq[n] = s;
}

// ---------------------------------------------------------------------------
// Kernel 2 (fast path, stage 1): one-time fp32 -> bf16 hi/lo split of the DB
// into WMMA A-operand chunk layout: blob[tile][hl][s][c][L] of 16B chunks.
// ---------------------------------------------------------------------------
__global__ __launch_bounds__(256) void presplit_kernel(const float* __restrict__ db,
                                                       uint4* __restrict__ blob) {
    const int gid  = blockIdx.x * 256 + threadIdx.x;   // one 16B chunk pair each
    const int tile = gid >> 10;
    const int g    = gid & 1023;
    const int r  = g >> 6;            // db row in tile
    const int kg = (g & 63) * 8;      // k offset (coalesced reads across threads)
    const int s  = kg >> 5;
    const int ks = kg & 31;
    const int c  = ks >> 4;
    const int h  = (ks >> 3) & 1;
    const int L  = h * 16 + r;
    const float4* p = (const float4*)(db + (size_t)(tile * TILE_M + r) * D_DIM + kg);
    float4 a = p[0], b = p[1];
    float f[8] = {a.x, a.y, a.z, a.w, b.x, b.y, b.z, b.w};
    uint4 hi, lo; split8(f, hi, lo);
    uint4* tb = blob + (size_t)tile * CHUNKS_PER_TILE;
    tb[((0 * NSLICES + s) * 2 + c) * 32 + L] = hi;
    tb[((1 * NSLICES + s) * 2 + c) * 32 + L] = lo;
}

// ---------------------------------------------------------------------------
// Kernel 3 (fast path, stage 2): per-segment WMMA distance scan.
//   - Query B-operands (loop-invariant) live entirely in VGPRs (256 regs).
//   - DB tiles stream through a 2x32KB LDS double buffer filled by
//     global_load_async_to_lds_b128 (overlapped with WMMA compute).
//   score[q][n] = ||d_n||^2 - 2 q.d_n   (||q||^2 rank-invariant; bf16x3 split)
// ---------------------------------------------------------------------------
__global__ __launch_bounds__(WAVES_PER_WG * 32) void knn_seg_kernel(
    const float* __restrict__ query,
    const char*  __restrict__ blob,      // staged bf16 hi/lo DB tiles
    const float* __restrict__ dsq,
    float* __restrict__ part_s,          // [Q][NSPLIT][16]
    int*   __restrict__ part_i)
{
    __shared__ uint4 sAbuf[2][2][NSLICES][2][32];          // 2 x 32 KB double buffer

    const int tid  = threadIdx.x;
    const int wave = tid >> 5;
    const int ln   = tid & 31;
    const int half = ln >> 4;
    const int qloc = ln & 15;
    const int seg   = blockIdx.x;
    const int qbase = blockIdx.y * Q_PER_WG + wave * Q_PER_WAVE;
    const int tile0 = seg * TPS;
    const unsigned ldsA = (unsigned)(unsigned long long)(void*)&sAbuf[0][0][0][0][0];

    // ---- Build this wave's 16-query B operands directly in registers.
    V16 bh[NSLICES], bl[NSLICES];
    {
        const float* qrow = query + (size_t)(qbase + qloc) * D_DIM;
        #pragma unroll
        for (int s = 0; s < NSLICES; ++s) {
            #pragma unroll
            for (int c = 0; c < 2; ++c) {
                const int k0 = s * 32 + c * 16 + half * 8;
                const float4* p = (const float4*)(qrow + k0);
                float4 a = p[0], b = p[1];
                float f[8] = {a.x, a.y, a.z, a.w, b.x, b.y, b.z, b.w};
                uint4 hi, lo; split8(f, hi, lo);
                bh[s].u[c] = hi;
                bl[s].u[c] = lo;
            }
        }
    }

    // ---- Prologue: async-copy first tile into buffer 0.
    {
        const char* src = blob + (size_t)tile0 * TILE_BYTES;
        #pragma unroll
        for (int i = 0; i < CHUNKS_PER_TILE / (WAVES_PER_WG * 32); ++i) {
            unsigned off = (unsigned)(i * (WAVES_PER_WG * 32) + tid) * 16u;
            async_copy_b128(ldsA + off, off, src);
        }
    }
    wait_async0();
    __syncthreads();

    float bd[KNN_K]; int bi[KNN_K];
    #pragma unroll
    for (int t = 0; t < KNN_K; ++t) { bd[t] = 3.0e38f; bi[t] = 0; }

    for (int t = 0; t < TPS; ++t) {
        const int cur  = t & 1;
        const int tile = tile0 + t;

        // ---- Issue async copy of next tile into the other buffer (no wait).
        if (t + 1 < TPS) {
            const char* src = blob + (size_t)(tile + 1) * TILE_BYTES;
            const unsigned dbase = ldsA + (unsigned)(cur ^ 1) * (unsigned)TILE_BYTES;
            #pragma unroll
            for (int i = 0; i < CHUNKS_PER_TILE / (WAVES_PER_WG * 32); ++i) {
                unsigned off = (unsigned)(i * (WAVES_PER_WG * 32) + tid) * 16u;
                async_copy_b128(dbase + off, off, src);
            }
        }

        // ---- 48 WMMAs per wave per tile: cross = qh.dh + qh.dl + ql.dh
        // Fully unrolled so bh[s]/bl[s] stay register-resident (constant index).
        v8f acc = {0.f, 0.f, 0.f, 0.f, 0.f, 0.f, 0.f, 0.f};
        #pragma unroll
        for (int s = 0; s < NSLICES; ++s) {
            V16 ah, al;
            ah.u[0] = sAbuf[cur][0][s][0][ln]; ah.u[1] = sAbuf[cur][0][s][1][ln];
            al.u[0] = sAbuf[cur][1][s][0][ln]; al.u[1] = sAbuf[cur][1][s][1][ln];
            acc = __builtin_amdgcn_wmma_f32_16x16x32_bf16(false, ah.v, false, bh[s].v,
                                                          (short)0, acc, false, false);
            acc = __builtin_amdgcn_wmma_f32_16x16x32_bf16(false, ah.v, false, bl[s].v,
                                                          (short)0, acc, false, false);
            acc = __builtin_amdgcn_wmma_f32_16x16x32_bf16(false, al.v, false, bh[s].v,
                                                          (short)0, acc, false, false);
        }

        // ---- Score + private sorted top-16 (1 compare/candidate common path)
        const float4* dq4 = (const float4*)(dsq + tile * TILE_M + half * 8);
        float4 d0 = dq4[0], d1 = dq4[1];
        float dq[8] = {d0.x, d0.y, d0.z, d0.w, d1.x, d1.y, d1.z, d1.w};
        const int mbase = tile * TILE_M + half * 8;
        #pragma unroll
        for (int i = 0; i < 8; ++i) {
            float sc = dq[i] - 2.0f * acc[i];
            if (sc < bd[KNN_K - 1]) {
                float cs = sc; int ci = mbase + i;
                #pragma unroll
                for (int u = 0; u < KNN_K; ++u) {
                    if (cs < bd[u]) {
                        float td = bd[u]; int ti = bi[u];
                        bd[u] = cs; bi[u] = ci; cs = td; ci = ti;
                    }
                }
            }
        }

        wait_async0();      // my share of next tile landed
        __syncthreads();    // everyone's share landed; reads of cur done
    }

    // ---- Merge lane L with lane L+16 per query; emit sorted segment top-16.
    float* ms = (float*)&sAbuf[0][0][0][0][0];           // reuse 64 KB buffer
    int*   mi = (int*)(ms + WAVES_PER_WG * 32 * KNN_K);
    const int slot = (wave * 32 + ln) * KNN_K;
    #pragma unroll
    for (int u = 0; u < KNN_K; ++u) { ms[slot + u] = bd[u]; mi[slot + u] = bi[u]; }
    __syncthreads();
    if (ln < 16) {
        const int xa = (wave * 32 + ln) * KNN_K;
        const int xb = (wave * 32 + ln + 16) * KNN_K;
        int i = 0, j = 0;
        const size_t row = (size_t)(qbase + ln) * NSPLIT + seg;
        float* ds_ = part_s + row * KNN_K;
        int*   di_ = part_i + row * KNN_K;
        #pragma unroll
        for (int u = 0; u < KNN_K; ++u) {       // i+j == u <= 15 -> in range
            float da = ms[xa + i], db_ = ms[xb + j];
            if (da <= db_) { ds_[u] = da;  di_[u] = mi[xa + i]; ++i; }
            else           { ds_[u] = db_; di_[u] = mi[xb + j]; ++j; }
        }
    }
}

// ---------------------------------------------------------------------------
// Kernel 4 (fast path, stage 3): merge NSPLIT sorted 16-lists per query.
// ---------------------------------------------------------------------------
__global__ __launch_bounds__(256) void knn_merge_kernel(const float* __restrict__ part_s,
                                                        const int* __restrict__ part_i,
                                                        int* __restrict__ topk) {
    const int q = blockIdx.x * 256 + threadIdx.x;
    float bd[KNN_K]; int bi[KNN_K];
    #pragma unroll
    for (int t = 0; t < KNN_K; ++t) { bd[t] = 3.0e38f; bi[t] = 0; }
    for (int s = 0; s < NSPLIT; ++s) {
        const size_t row = ((size_t)q * NSPLIT + s) * KNN_K;
        for (int t = 0; t < KNN_K; ++t) {
            float sc = part_s[row + t];
            if (sc >= bd[KNN_K - 1]) break;      // segment list is sorted ascending
            float cs = sc; int ci = part_i[row + t];
            #pragma unroll
            for (int u = 0; u < KNN_K; ++u) {
                if (cs < bd[u]) {
                    float td = bd[u]; int ti = bi[u];
                    bd[u] = cs; bi[u] = ci; cs = td; ci = ti;
                }
            }
        }
    }
    #pragma unroll
    for (int t = 0; t < KNN_K; ++t) topk[(size_t)q * KNN_K + t] = bi[t];
}

// ---------------------------------------------------------------------------
// Fallback (small workspace): fused kernel with in-loop fp32->bf16 staging
// (round-1 version).
// ---------------------------------------------------------------------------
__global__ __launch_bounds__(WAVES_PER_WG * 32) void knn_topk_fallback(
    const float* __restrict__ query,
    const float* __restrict__ database,
    const float* __restrict__ dsq,
    int* __restrict__ topk)
{
    __shared__ uint4 sA[2][NSLICES][2][32];
    __shared__ uint4 sQ[WAVES_PER_WG][2][NSLICES][2][32];

    const int tid  = threadIdx.x;
    const int wave = tid >> 5;
    const int ln   = tid & 31;
    const int half = ln >> 4;
    const int qloc = ln & 15;
    const int qbase = blockIdx.x * Q_PER_WG + wave * Q_PER_WAVE;

    {
        const float* qrow = query + (size_t)(qbase + qloc) * D_DIM;
        #pragma unroll
        for (int s = 0; s < NSLICES; ++s) {
            #pragma unroll
            for (int c = 0; c < 2; ++c) {
                const int k0 = s * 32 + c * 16 + half * 8;
                const float4* p = (const float4*)(qrow + k0);
                float4 a = p[0], b = p[1];
                float f[8] = {a.x, a.y, a.z, a.w, b.x, b.y, b.z, b.w};
                uint4 hi, lo; split8(f, hi, lo);
                sQ[wave][0][s][c][ln] = hi;
                sQ[wave][1][s][c][ln] = lo;
            }
        }
    }

    float bd[KNN_K]; int bi[KNN_K];
    #pragma unroll
    for (int t = 0; t < KNN_K; ++t) { bd[t] = 3.0e38f; bi[t] = 0; }

    for (int tile = 0; tile < NTILES; ++tile) {
        __syncthreads();
        for (int g = tid; g < 1024; g += WAVES_PER_WG * 32) {
            const int r  = g >> 6;
            const int kg = (g & 63) * 8;
            const int s  = kg >> 5;
            const int ks = kg & 31;
            const int c  = ks >> 4;
            const int h  = (ks >> 3) & 1;
            const int L  = h * 16 + r;
            const float4* p = (const float4*)(database + (size_t)(tile * TILE_M + r) * D_DIM + kg);
            float4 a = p[0], b = p[1];
            float f[8] = {a.x, a.y, a.z, a.w, b.x, b.y, b.z, b.w};
            uint4 hi, lo; split8(f, hi, lo);
            sA[0][s][c][L] = hi;
            sA[1][s][c][L] = lo;
        }
        if (tile + 1 < NTILES)
            __builtin_prefetch(database + (size_t)((tile + 1) * TILE_M + (tid & 15)) * D_DIM
                                        + (tid >> 4) * 16, 0, 1);
        __syncthreads();

        v8f acc = {0.f, 0.f, 0.f, 0.f, 0.f, 0.f, 0.f, 0.f};
        #pragma unroll 4
        for (int s = 0; s < NSLICES; ++s) {
            V16 ah, al, bh, bl;
            ah.u[0] = sA[0][s][0][ln]; ah.u[1] = sA[0][s][1][ln];
            al.u[0] = sA[1][s][0][ln]; al.u[1] = sA[1][s][1][ln];
            bh.u[0] = sQ[wave][0][s][0][ln]; bh.u[1] = sQ[wave][0][s][1][ln];
            bl.u[0] = sQ[wave][1][s][0][ln]; bl.u[1] = sQ[wave][1][s][1][ln];
            acc = __builtin_amdgcn_wmma_f32_16x16x32_bf16(false, ah.v, false, bh.v,
                                                          (short)0, acc, false, false);
            acc = __builtin_amdgcn_wmma_f32_16x16x32_bf16(false, ah.v, false, bl.v,
                                                          (short)0, acc, false, false);
            acc = __builtin_amdgcn_wmma_f32_16x16x32_bf16(false, al.v, false, bh.v,
                                                          (short)0, acc, false, false);
        }

        const float4* dq4 = (const float4*)(dsq + tile * TILE_M + half * 8);
        float4 d0 = dq4[0], d1 = dq4[1];
        float dq[8] = {d0.x, d0.y, d0.z, d0.w, d1.x, d1.y, d1.z, d1.w};
        const int mbase = tile * TILE_M + half * 8;
        #pragma unroll
        for (int i = 0; i < 8; ++i) {
            float sc = dq[i] - 2.0f * acc[i];
            if (sc < bd[KNN_K - 1]) {
                float cs = sc; int ci = mbase + i;
                #pragma unroll
                for (int u = 0; u < KNN_K; ++u) {
                    if (cs < bd[u]) {
                        float td = bd[u]; int ti = bi[u];
                        bd[u] = cs; bi[u] = ci; cs = td; ci = ti;
                    }
                }
            }
        }
    }

    __syncthreads();
    float* ms = (float*)&sA[0][0][0][0];
    int*   mi = (int*)(ms + WAVES_PER_WG * 32 * KNN_K);
    const int slot = (wave * 32 + ln) * KNN_K;
    #pragma unroll
    for (int t = 0; t < KNN_K; ++t) { ms[slot + t] = bd[t]; mi[slot + t] = bi[t]; }
    __syncthreads();
    if (ln < 16) {
        const int xa = (wave * 32 + ln) * KNN_K;
        const int xb = (wave * 32 + ln + 16) * KNN_K;
        int i = 0, j = 0;
        int* dst = topk + (size_t)(qbase + ln) * KNN_K;
        #pragma unroll
        for (int t = 0; t < KNN_K; ++t) {
            float da = ms[xa + i], db_ = ms[xb + j];
            if (da <= db_) { dst[t] = mi[xa + i]; ++i; }
            else           { dst[t] = mi[xb + j]; ++j; }
        }
    }
}

// ---------------------------------------------------------------------------
// Final kernel: gather labels of the 16 neighbors and average.
// ---------------------------------------------------------------------------
__global__ __launch_bounds__(C_DIM) void labels_kernel(const int* __restrict__ topk,
                                                       const float* __restrict__ labels,
                                                       float* __restrict__ out) {
    __shared__ int idx[KNN_K];
    const int q = blockIdx.x, c = threadIdx.x;
    if (c < KNN_K) idx[c] = topk[q * KNN_K + c];
    __syncthreads();
    float s = 0.f;
    #pragma unroll
    for (int j = 0; j < KNN_K; ++j) s += labels[(size_t)idx[j] * C_DIM + c];
    out[(size_t)q * C_DIM + c] = s * (1.0f / (float)KNN_K);
}

// ---------------------------------------------------------------------------
extern "C" void kernel_launch(void* const* d_in, const int* in_sizes, int n_in,
                              void* d_out, int out_size, void* d_ws, size_t ws_size,
                              hipStream_t stream) {
    (void)in_sizes; (void)n_in; (void)out_size;
    const float* query    = (const float*)d_in[0];   // [4096, 512]
    const float* database = (const float*)d_in[1];   // [65536, 512]
    const float* labels   = (const float*)d_in[2];   // [65536, 128]
    float* out = (float*)d_out;                      // [4096, 128]

    // Workspace layout
    char* ws = (char*)d_ws;
    int*   topk = (int*)ws;                                       // 256 KB @ 0
    float* dsq  = (float*)(ws + 262144);                          // 256 KB @ 256K
    float* part_s = (float*)(ws + 524288);                        // 2 MB   @ 512K
    int*   part_i = (int*)(ws + 524288 + 2097152);                // 2 MB
    char*  blob   = ws + 4718592;                                 // 128 MB staged DB
    const size_t need_fast = 4718592ull + (size_t)NTILES * TILE_BYTES;

    dbnorm_kernel<<<N_DB / 8, 256, 0, stream>>>(database, dsq);

    if (ws_size >= need_fast) {
        presplit_kernel<<<(NTILES * 1024) / 256, 256, 0, stream>>>(database, (uint4*)blob);
        knn_seg_kernel<<<dim3(NSPLIT, Q_TOTAL / Q_PER_WG), WAVES_PER_WG * 32, 0, stream>>>(
            query, blob, dsq, part_s, part_i);
        knn_merge_kernel<<<Q_TOTAL / 256, 256, 0, stream>>>(part_s, part_i, topk);
    } else {
        knn_topk_fallback<<<Q_TOTAL / Q_PER_WG, WAVES_PER_WG * 32, 0, stream>>>(
            query, database, dsq, topk);
    }

    labels_kernel<<<Q_TOTAL, C_DIM, 0, stream>>>(topk, labels, out);
}
